// KerasDense_52596169506952
// MI455X (gfx1250) — matrix operand order repaired and checked by reading, not compiled, not run on hardware
//
#include <hip/hip_runtime.h>

// ---------------------------------------------------------------------------
// TT-matrix times dense batch on gfx1250 (MI455X, wave32, WMMA).
// Strategy: materialize dense W^T (bf16, 32MB, L2-resident) from the 4 TT
// cores with cheap scalar kernels, convert x to bf16, then run a
// v_wmma_f32_16x16x32_bf16 GEMM with fused bias+relu.
// Wave tile raised to 64x64 (4x4 accumulators) -> 1.0 VMEM per WMMA.
// ---------------------------------------------------------------------------

typedef __attribute__((ext_vector_type(16))) __bf16        v16bf;
typedef __attribute__((ext_vector_type(8)))  float         v8f;
typedef __attribute__((ext_vector_type(4)))  unsigned int  v4u;

#define N_DIM 4096
#define K_DIM 4096

union Frag {
    v16bf v;
    v4u   u[2];
};

__device__ __forceinline__ unsigned short f32_bf16(float f) {
    unsigned int u = __float_as_uint(f);
    unsigned int r = u + 0x7FFFu + ((u >> 16) & 1u);   // round-to-nearest-even
    return (unsigned short)(r >> 16);
}

// ---- Stage 1: W12[(i1 i2),(j1 j2),r2] = sum_r1 G0[i1,j1,r1]*G1[r1,i2,j2,r2]
__global__ __launch_bounds__(256) void tt_stage1(
        const float* __restrict__ c0, const float* __restrict__ c1,
        float* __restrict__ w12) {
    int idx = blockIdx.x * 256 + threadIdx.x;          // 64*64*16 = 65536
    int r2 = idx & 15;
    int t  = idx >> 4;
    int j12 = t & 63, i12 = t >> 6;
    int i1 = i12 >> 3, i2 = i12 & 7, j1 = j12 >> 3, j2 = j12 & 7;
    float s = 0.f;
#pragma unroll
    for (int r1 = 0; r1 < 16; ++r1)
        s += c0[(i1 * 8 + j1) * 16 + r1] *
             c1[(((r1 * 8 + i2) * 8 + j2) * 16) + r2];
    w12[idx] = s;
}

// ---- Stage 2: W123[(i123),(j123),r3] = sum_r2 W12 * G2[r2,i3,j3,r3]
__global__ __launch_bounds__(256) void tt_stage2(
        const float* __restrict__ w12, const float* __restrict__ c2,
        float* __restrict__ w123) {
    int idx = blockIdx.x * 256 + threadIdx.x;          // 512*512*16 = 4194304
    int r3 = idx & 15;
    int t  = idx >> 4;
    int j123 = t & 511, i123 = t >> 9;
    int i12 = i123 >> 3, i3 = i123 & 7, j12 = j123 >> 3, j3 = j123 & 7;
    float s = 0.f;
#pragma unroll
    for (int r2 = 0; r2 < 16; ++r2)
        s += w12[((i12 * 64 + j12) * 16) + r2] *
             c2[(((r2 * 8 + i3) * 8 + j3) * 16) + r3];
    w123[idx] = s;
}

// ---- Stage 3: Wt[j][i] (bf16, K-contiguous per column) = sum_r3 W123 * G3
__global__ __launch_bounds__(256) void tt_stage3(
        const float* __restrict__ w123, const float* __restrict__ c3,
        unsigned short* __restrict__ wt) {
    int idx = blockIdx.x * 256 + threadIdx.x;          // 4096*4096, idx=j*4096+i
    int i = idx & 4095, j = idx >> 12;
    int i123 = i >> 3, i4 = i & 7, j123 = j >> 3, j4 = j & 7;
    float s = 0.f;
#pragma unroll
    for (int r3 = 0; r3 < 16; ++r3)
        s += w123[((i123 * 512 + j123) * 16) + r3] *
             c3[(r3 * 8 + i4) * 8 + j4];
    wt[idx] = f32_bf16(s);
}

// ---- Convert x (fp32 row-major) -> bf16
__global__ __launch_bounds__(256) void conv_x(
        const float* __restrict__ x, unsigned short* __restrict__ xb, int n) {
    int idx = blockIdx.x * 256 + threadIdx.x;
    if (idx < n) xb[idx] = f32_bf16(x[idx]);
}

// ---- WMMA GEMM: out = relu(xb * Wt^T + bias), fp32 output.
// Block = 256 threads = 8 waves, block tile 128(M) x 256(N),
// per-wave tile 64(M) x 64(N) -> 4x4 accumulators of v8f.
__global__ __launch_bounds__(256) void tt_gemm_wmma(
        const unsigned short* __restrict__ xb,   // [batch][4096] bf16 row-major
        const unsigned short* __restrict__ wt,   // [4096 N][4096 K] bf16 (K contig)
        const float* __restrict__ bias,          // [4096]
        float* __restrict__ out) {               // [batch][4096] fp32
    const int tid   = threadIdx.x;
    const int lane  = tid & 31;
    const int wave  = tid >> 5;
    const int waveM = wave & 1;       // 2 waves along M (64 rows each)
    const int waveN = wave >> 1;      // 4 waves along N (64 cols each)
    const int blockN0 = blockIdx.x * 256;
    const int blockM0 = blockIdx.y * 128;
    const int half  = (lane < 16) ? 0 : 1;
    const int l15   = lane & 15;

    // A-fragment per-lane K chunk: lane<16 -> K 0..15; lane>=16 -> K 16..31
    const int aOff = half ? 16 : 0;
    // B-fragment per-lane K chunk: lane<16 -> K 0..15; lane>=16 -> K 16..31
    const int bOff = half ? 16 : 0;

    const int rowBase = blockM0 + waveM * 64 + l15;   // + mt*16 = A row (M=lane&15)
    const int colBase = blockN0 + waveN * 64 + l15;   // + nt*16 = B col (N=lane&15)

    const unsigned short* aPtr[4];
#pragma unroll
    for (int mt = 0; mt < 4; ++mt)
        aPtr[mt] = xb + (size_t)(rowBase + mt * 16) * K_DIM + aOff;
    const unsigned short* bPtr[4];
#pragma unroll
    for (int nt = 0; nt < 4; ++nt)
        bPtr[nt] = wt + (size_t)(colBase + nt * 16) * K_DIM + bOff;

    v8f acc[4][4] = {};

    for (int kb = 0; kb < K_DIM; kb += 32) {
        Frag a[4], b[4];
#pragma unroll
        for (int mt = 0; mt < 4; ++mt) {
            a[mt].u[0] = *(const v4u*)(aPtr[mt] + kb);        // K = aOff+0..7
            a[mt].u[1] = *(const v4u*)(aPtr[mt] + kb + 8);    // K = aOff+8..15
        }
#pragma unroll
        for (int nt = 0; nt < 4; ++nt) {
            b[nt].u[0] = *(const v4u*)(bPtr[nt] + kb);        // K = bOff+0..7
            b[nt].u[1] = *(const v4u*)(bPtr[nt] + kb + 8);    // K = bOff+8..15
        }
#pragma unroll
        for (int mt = 0; mt < 4; ++mt)
#pragma unroll
            for (int nt = 0; nt < 4; ++nt)
                acc[mt][nt] = __builtin_amdgcn_wmma_f32_16x16x32_bf16(
                    false, a[mt].v, false, b[nt].v,
                    (short)0, acc[mt][nt], false, false);
    }

    // Epilogue: C/D layout -> VGPR r holds (M = r + 8*half, N = lane&15)
#pragma unroll
    for (int mt = 0; mt < 4; ++mt) {
        const int row0 = blockM0 + waveM * 64 + mt * 16 + half * 8;
#pragma unroll
        for (int nt = 0; nt < 4; ++nt) {
            const int col = colBase + nt * 16;
            const float bv = bias[col];
#pragma unroll
            for (int r = 0; r < 8; ++r) {
                float v = acc[mt][nt][r] + bv;
                out[(size_t)(row0 + r) * N_DIM + col] = v > 0.f ? v : 0.f;
            }
        }
    }
}

extern "C" void kernel_launch(void* const* d_in, const int* in_sizes, int n_in,
                              void* d_out, int out_size, void* d_ws, size_t ws_size,
                              hipStream_t stream) {
    const float* x     = (const float*)d_in[0];
    const float* core0 = (const float*)d_in[1];
    const float* core1 = (const float*)d_in[2];
    const float* core2 = (const float*)d_in[3];
    const float* core3 = (const float*)d_in[4];
    const float* bias  = (const float*)d_in[5];
    float* out = (float*)d_out;

    const int batch = in_sizes[0] / K_DIM;   // 8192
    const int xElems = batch * K_DIM;

    // Workspace layout (all 16B-aligned):
    //   w12  : 64*64*16 f32        =   256 KB
    //   w123 : 512*512*16 f32      =    16 MB
    //   wt   : 4096*4096 bf16      =    32 MB
    //   xb   : batch*4096 bf16     =    64 MB
    char* ws = (char*)d_ws;
    float*          w12  = (float*)(ws);
    float*          w123 = (float*)(ws + (size_t)256 * 1024);
    unsigned short* wt   = (unsigned short*)(ws + (size_t)256 * 1024 + (size_t)4194304 * 4);
    unsigned short* xb   = (unsigned short*)(ws + (size_t)256 * 1024 + (size_t)4194304 * 4
                                                + (size_t)16777216 * 2);

    tt_stage1<<<65536 / 256, 256, 0, stream>>>(core0, core1, w12);
    tt_stage2<<<4194304 / 256, 256, 0, stream>>>(w12, core2, w123);
    tt_stage3<<<16777216 / 256, 256, 0, stream>>>(w123, core3, wt);
    conv_x<<<(xElems + 255) / 256, 256, 0, stream>>>(x, xb, xElems);

    dim3 grid(N_DIM / 256, batch / 128);   // 16 x 64 blocks
    tt_gemm_wmma<<<grid, 256, 0, stream>>>(xb, wt, bias, out);
}
